// Mamba_49847390437667
// MI455X (gfx1250) — matrix operand order repaired and checked by reading, not compiled, not run on hardware
//
#include <hip/hip_runtime.h>
#include <hip/hip_bf16.h>
#include <stdint.h>

// ---------------------------------------------------------------------------
// Types for CDNA5 WMMA (wave32): V_WMMA_F32_16X16X32_BF16
// ---------------------------------------------------------------------------
typedef __attribute__((ext_vector_type(16))) __bf16    v16bf;
typedef __attribute__((ext_vector_type(8)))  float     v8f;
typedef __attribute__((ext_vector_type(8)))  uint16_t  u16x8;
typedef __attribute__((ext_vector_type(16))) uint16_t  u16x16;

union BfVec16 { u16x16 u; v16bf h; };

__device__ __forceinline__ uint16_t f32_to_bf16_rne(float f) {
  uint32_t u = __float_as_uint(f);
  u += 0x7FFFu + ((u >> 16) & 1u);         // round-to-nearest-even
  return (uint16_t)(u >> 16);
}

// Load one 16x32 bf16 A-fragment (ISA 7.12.2 lane layout) from a row pointer.
__device__ __forceinline__ v16bf load_a_frag(const uint16_t* arow, int k,
                                             int ka0, int ka1) {
  u16x8 a0 = *(const u16x8*)(arow + k + ka0);
  u16x8 a1 = *(const u16x8*)(arow + k + ka1);
  BfVec16 av;
  av.u = __builtin_shufflevector(a0, a1, 0,1,2,3,4,5,6,7,8,9,10,11,12,13,14,15);
  return av.h;
}

// Load one 32x16 bf16 B-fragment (same k order as A: kb0..kb0+7, kb0+16..kb0+23).
__device__ __forceinline__ v16bf load_b_frag(const uint16_t* brow, int k, int kb0) {
  u16x8 b0 = *(const u16x8*)(brow + k + kb0);
  u16x8 b1 = *(const u16x8*)(brow + k + kb0 + 16);
  BfVec16 bv;
  bv.u = __builtin_shufflevector(b0, b1, 0,1,2,3,4,5,6,7,8,9,10,11,12,13,14,15);
  return bv.h;
}

// ---------------------------------------------------------------------------
// Elementwise fp32 -> bf16 converter (for weights/activations feeding WMMA)
// ---------------------------------------------------------------------------
__global__ void mamba_cvt_bf16(const float* __restrict__ in,
                               uint16_t* __restrict__ out, int n) {
  int i = blockIdx.x * blockDim.x + threadIdx.x;
  if (i < n) out[i] = f32_to_bf16_rne(in[i]);
}

// ---------------------------------------------------------------------------
// GEMM: C[M,N] = A[M,K](bf16, row stride lda) * W[N,K](bf16)^T + bias
// act: 0 = none, 1 = softplus.
// One wave -> 32x32 C macro-tile (2x2 grid of 16x16 WMMA tiles): per k-step
// loads 2 A-frags + 2 B-frags and issues 4 independent WMMAs (halves L2
// traffic per FLOP vs 1x1 tiling; 4 parallel accumulation chains).
// ---------------------------------------------------------------------------
__global__ __launch_bounds__(256)
void mamba_gemm_bf16_wmma(const uint16_t* __restrict__ A,
                          const uint16_t* __restrict__ W,
                          const float* __restrict__ bias,
                          float* __restrict__ C,
                          int M, int N, int K, int lda, int act) {
  const int wave   = blockIdx.x * (blockDim.x >> 5) + (threadIdx.x >> 5);
  const int tilesN = N >> 5;                    // 32-wide macro tiles
  const int total  = (M >> 5) * tilesN;
  if (wave >= total) return;
  const int tm   = wave / tilesN;
  const int tn   = wave % tilesN;
  const int lane = threadIdx.x & 31;
  const int r16  = lane & 15;
  const int half = lane >> 4;

  const uint16_t* arow0 = A + (size_t)(tm * 32 + r16) * lda;
  const uint16_t* arow1 = arow0 + (size_t)16 * lda;
  const uint16_t* brow0 = W + (size_t)(tn * 32 + r16) * K;
  const uint16_t* brow1 = brow0 + (size_t)16 * K;
  const int ka0 = half * 8;
  const int ka1 = 16 + half * 8;
  const int kb0 = half * 8;

  v8f acc00 = {}, acc01 = {}, acc10 = {}, acc11 = {};
  for (int k = 0; k < K; k += 32) {
    v16bf a0 = load_a_frag(arow0, k, ka0, ka1);
    v16bf a1 = load_a_frag(arow1, k, ka0, ka1);
    v16bf b0 = load_b_frag(brow0, k, kb0);
    v16bf b1 = load_b_frag(brow1, k, kb0);
    acc00 = __builtin_amdgcn_wmma_f32_16x16x32_bf16(false, a0, false, b0, (short)0, acc00, false, false);
    acc01 = __builtin_amdgcn_wmma_f32_16x16x32_bf16(false, a0, false, b1, (short)0, acc01, false, false);
    acc10 = __builtin_amdgcn_wmma_f32_16x16x32_bf16(false, a1, false, b0, (short)0, acc10, false, false);
    acc11 = __builtin_amdgcn_wmma_f32_16x16x32_bf16(false, a1, false, b1, (short)0, acc11, false, false);
  }

  // C/D layout: vgpr r -> row (half*8 + r), col (lane & 15)
  const float bv0 = bias ? bias[tn * 32 + r16] : 0.0f;
  const float bv1 = bias ? bias[tn * 32 + 16 + r16] : 0.0f;
  const int rb0 = tm * 32 + half * 8;
  const int col0 = tn * 32 + r16;
  #pragma unroll
  for (int r = 0; r < 8; ++r) {
    float v00 = acc00[r] + bv0;
    float v01 = acc01[r] + bv1;
    float v10 = acc10[r] + bv0;
    float v11 = acc11[r] + bv1;
    if (act == 1) {
      v00 = (v00 > 20.0f) ? v00 : log1pf(expf(v00));
      v01 = (v01 > 20.0f) ? v01 : log1pf(expf(v01));
      v10 = (v10 > 20.0f) ? v10 : log1pf(expf(v10));
      v11 = (v11 > 20.0f) ? v11 : log1pf(expf(v11));
    }
    C[(size_t)(rb0 + r) * N + col0]           = v00;
    C[(size_t)(rb0 + r) * N + col0 + 16]      = v01;
    C[(size_t)(rb0 + 16 + r) * N + col0]      = v10;
    C[(size_t)(rb0 + 16 + r) * N + col0 + 16] = v11;
  }
}

// ---------------------------------------------------------------------------
// Causal depthwise conv1d (d_conv=4) + bias + SiLU.
// Reads x half of xz (B,L,4096); writes fp32 x_act and bf16 copy.
// ---------------------------------------------------------------------------
__global__ __launch_bounds__(256)
void mamba_conv_silu(const float* __restrict__ xz,
                     const float* __restrict__ conv_w,
                     const float* __restrict__ conv_b,
                     float* __restrict__ x_act,
                     uint16_t* __restrict__ x_act_bf, int total) {
  int idx = blockIdx.x * blockDim.x + threadIdx.x;
  if (idx >= total) return;
  const int DIN = 2048, L = 1024;
  int d = idx & (DIN - 1);
  int l = (idx / DIN) & (L - 1);
  int b = idx / (DIN * L);
  float acc = conv_b[d];
  #pragma unroll
  for (int j = 0; j < 4; ++j) {
    int ll = l - 3 + j;
    if (ll >= 0)
      acc += xz[((size_t)(b * L + ll)) * 4096 + d] * conv_w[d * 4 + j];
  }
  float s = acc / (1.0f + __expf(-acc));   // silu
  x_act[idx]    = s;
  x_act_bf[idx] = f32_to_bf16_rne(s);
}

// ---------------------------------------------------------------------------
// Selective scan. One thread per (batch, channel d): 16-state recurrence in
// registers; B_t/C_t (shared across all d) staged via LDS in chunks of 8
// timesteps. Fuses the D-residual and silu(z) gate; emits bf16 y for the
// out_proj WMMA GEMM.
// ---------------------------------------------------------------------------
__global__ __launch_bounds__(256)
void mamba_scan(const float* __restrict__ x_dbl,   // (B*L, 96): [dt_lr|B|C]
                const float* __restrict__ dt,      // (B*L, 2048) softplus'ed
                const float* __restrict__ x_act,   // (B*L, 2048)
                const float* __restrict__ xz,      // (B*L, 4096), z at +2048
                const float* __restrict__ A_log,   // (2048, 16)
                const float* __restrict__ D_param, // (2048)
                uint16_t* __restrict__ y_bf) {     // (B*L, 2048)
  const int L = 1024, DIN = 2048, NS = 16, XPN = 96;
  const int tid = threadIdx.x;
  const int d = blockIdx.x * blockDim.x + tid;
  const int b = blockIdx.y;

  float Aa[NS], h[NS];
  #pragma unroll
  for (int n = 0; n < NS; ++n) {
    Aa[n] = -__expf(A_log[d * NS + n]);
    h[n] = 0.0f;
  }
  const float Dv = D_param[d];

  __shared__ float Bs[8][NS];
  __shared__ float Cs[8][NS];

  const int jload = tid >> 5;        // 0..7 timestep in chunk
  const int iload = tid & 31;        // 0..31 -> 16 B + 16 C

  for (int l0 = 0; l0 < L; l0 += 8) {
    float v = x_dbl[(size_t)(b * L + l0 + jload) * XPN + 64 + iload];
    if (iload < NS) Bs[jload][iload] = v;
    else            Cs[jload][iload - NS] = v;
    __syncthreads();

    for (int j = 0; j < 8; ++j) {
      const size_t m = (size_t)(b * L + l0 + j);
      const float dtv = dt[m * DIN + d];
      const float xv  = x_act[m * DIN + d];
      const float dtx = dtv * xv;
      float y = 0.0f;
      #pragma unroll
      for (int n = 0; n < NS; ++n) {
        float dA = __expf(dtv * Aa[n]);
        h[n] = dA * h[n] + dtx * Bs[j][n];
        y += h[n] * Cs[j][n];
      }
      const float zv = xz[m * 4096 + 2048 + d];
      const float g  = zv / (1.0f + __expf(-zv));  // silu(z)
      y_bf[m * DIN + d] = f32_to_bf16_rne((y + xv * Dv) * g);
    }
    __syncthreads();
  }
}

// ---------------------------------------------------------------------------
// Host-side orchestration
// ---------------------------------------------------------------------------
extern "C" void kernel_launch(void* const* d_in, const int* in_sizes, int n_in,
                              void* d_out, int out_size, void* d_ws, size_t ws_size,
                              hipStream_t stream) {
  constexpr int Bq = 2, L = 1024, DM = 1024, DIN = 2048, NS = 16, DTR = 64;
  constexpr int XPN = DTR + 2 * NS;           // 96
  constexpr int MR  = Bq * L;                 // 2048 rows for all GEMMs

  const float* hs       = (const float*)d_in[0];   // (B,L,DM)
  const float* w_in     = (const float*)d_in[1];   // (2*DIN, DM)
  const float* b_in     = (const float*)d_in[2];   // (2*DIN)
  const float* conv_w   = (const float*)d_in[3];   // (DIN, 4)
  const float* conv_b   = (const float*)d_in[4];   // (DIN)
  const float* w_xp     = (const float*)d_in[5];   // (XPN, DIN)
  const float* w_dt     = (const float*)d_in[6];   // (DIN, DTR)
  const float* b_dt     = (const float*)d_in[7];   // (DIN)
  const float* A_log    = (const float*)d_in[8];   // (DIN, NS)
  const float* D_param  = (const float*)d_in[9];   // (DIN)
  const float* w_out    = (const float*)d_in[10];  // (DM, DIN)
  const float* b_out    = (const float*)d_in[11];  // (DM)
  float* out = (float*)d_out;                      // (B,L,DM)

  // Workspace carve-up (256B aligned)
  char* ws = (char*)d_ws;
  size_t off = 0;
  auto carve = [&](size_t bytes) -> void* {
    void* p = ws + off;
    off = (off + bytes + 255) & ~(size_t)255;
    return p;
  };
  uint16_t* hs_bf    = (uint16_t*)carve((size_t)MR * DM * 2);
  uint16_t* w_in_bf  = (uint16_t*)carve((size_t)2 * DIN * DM * 2);
  uint16_t* w_xp_bf  = (uint16_t*)carve((size_t)XPN * DIN * 2);
  uint16_t* w_dt_bf  = (uint16_t*)carve((size_t)DIN * DTR * 2);
  uint16_t* w_out_bf = (uint16_t*)carve((size_t)DM * DIN * 2);
  float*    xz       = (float*)   carve((size_t)MR * 2 * DIN * 4);
  float*    x_act    = (float*)   carve((size_t)MR * DIN * 4);
  uint16_t* x_act_bf = (uint16_t*)carve((size_t)MR * DIN * 2);
  float*    x_dbl    = (float*)   carve((size_t)MR * XPN * 4);
  uint16_t* x_dbl_bf = (uint16_t*)carve((size_t)MR * XPN * 4); // padded
  float*    dtbuf    = (float*)   carve((size_t)MR * DIN * 4);
  uint16_t* y_bf     = (uint16_t*)carve((size_t)MR * DIN * 2);
  (void)ws_size; (void)in_sizes; (void)n_in; (void)out_size;

  auto cdiv = [](int a, int b) { return (a + b - 1) / b; };
  const dim3 blk(256);

  // 1) fp32 -> bf16 conversions for WMMA operands
  mamba_cvt_bf16<<<cdiv(MR * DM, 256), blk, 0, stream>>>(hs, hs_bf, MR * DM);
  mamba_cvt_bf16<<<cdiv(2 * DIN * DM, 256), blk, 0, stream>>>(w_in, w_in_bf, 2 * DIN * DM);
  mamba_cvt_bf16<<<cdiv(XPN * DIN, 256), blk, 0, stream>>>(w_xp, w_xp_bf, XPN * DIN);
  mamba_cvt_bf16<<<cdiv(DIN * DTR, 256), blk, 0, stream>>>(w_dt, w_dt_bf, DIN * DTR);
  mamba_cvt_bf16<<<cdiv(DM * DIN, 256), blk, 0, stream>>>(w_out, w_out_bf, DM * DIN);

  // 2) in_proj: xz(MR,4096) = hs(MR,1024) @ w_in^T + b_in
  {
    int tiles = (MR / 32) * ((2 * DIN) / 32);
    mamba_gemm_bf16_wmma<<<cdiv(tiles, 8), blk, 0, stream>>>(
        hs_bf, w_in_bf, b_in, xz, MR, 2 * DIN, DM, DM, 0);
  }

  // 3) causal depthwise conv + silu (emits fp32 + bf16)
  mamba_conv_silu<<<cdiv(MR * DIN, 256), blk, 0, stream>>>(
      xz, conv_w, conv_b, x_act, x_act_bf, MR * DIN);

  // 4) x_proj: x_dbl(MR,96) = x_act @ w_xp^T
  {
    int tiles = (MR / 32) * (XPN / 32);
    mamba_gemm_bf16_wmma<<<cdiv(tiles, 8), blk, 0, stream>>>(
        x_act_bf, w_xp_bf, nullptr, x_dbl, MR, XPN, DIN, DIN, 0);
  }
  mamba_cvt_bf16<<<cdiv(MR * XPN, 256), blk, 0, stream>>>(x_dbl, x_dbl_bf, MR * XPN);

  // 5) dt = softplus(dt_lr @ w_dt^T + b_dt); dt_lr = x_dbl[:, :64] (lda=96)
  {
    int tiles = (MR / 32) * (DIN / 32);
    mamba_gemm_bf16_wmma<<<cdiv(tiles, 8), blk, 0, stream>>>(
        x_dbl_bf, w_dt_bf, b_dt, dtbuf, MR, DIN, DTR, XPN, 1);
  }

  // 6) selective scan + D residual + silu(z) gate -> y_bf
  mamba_scan<<<dim3(DIN / 256, Bq), blk, 0, stream>>>(
      x_dbl, dtbuf, x_act, xz, A_log, D_param, y_bf);

  // 7) out_proj: out(MR,1024) = y @ w_out^T + b_out
  {
    int tiles = (MR / 32) * (DM / 32);
    mamba_gemm_bf16_wmma<<<cdiv(tiles, 8), blk, 0, stream>>>(
        y_bf, w_out_bf, b_out, out, MR, DM, DIN, DIN, 0);
  }
}
